// SlowRNN_83485574300263
// MI455X (gfx1250) — matrix operand order repaired and checked by reading, not compiled, not run on hardware
//
#include <hip/hip_runtime.h>
#include <hip/hip_bf16.h>

// ---------------------------------------------------------------------------
// SlowRNN on MI455X (gfx1250): bf16 WMMA (f32 accum) everywhere.
//   Phase 0: swizzle Wxh / Whh (f32 -> bf16) into the exact WMMA B-operand
//            fragment layout so each lane's fragment is contiguous.
//   Phase 1: xproj = x @ Wxh^T + bh  (big parallel GEMM, bf16 WMMA)
//   Phase 2: 4 persistent workgroups (one per 16-batch chunk), h in LDS,
//            512 sequential steps: h = tanh(xproj[t] + h @ Whh^T),
//            with global_prefetch of xproj[t+1] overlapping step t compute.
// ---------------------------------------------------------------------------

typedef __attribute__((ext_vector_type(16))) __bf16 v16bf;
typedef __attribute__((ext_vector_type(8)))  __bf16 v8bf;
typedef __attribute__((ext_vector_type(8)))  float  v8f;
typedef __attribute__((ext_vector_type(4)))  float  v4f;

constexpr int T_STEPS = 512;
constexpr int BATCH   = 64;
constexpr int DIN     = 1024;
constexpr int DH      = 1024;
constexpr int MTOT    = T_STEPS * BATCH;   // 32768 rows for phase-1 GEMM
constexpr int NT      = DH / 16;           // 64 n-tiles
constexpr int KT      = DH / 32;           // 32 k-tiles (K = 1024)

static __device__ __forceinline__ v8f wmma_bf16(v16bf a, v16bf b, v8f c) {
  // v_wmma_f32_16x16x32_bf16: D = A(16x32) * B(32x16) + C(16x16 f32)
  return __builtin_amdgcn_wmma_f32_16x16x32_bf16(
      /*neg_a=*/false, a, /*neg_b=*/false, b,
      /*c_mod=*/(short)0, c, /*reuse_a=*/false, /*reuse_b=*/false);
}

// ---------------------------------------------------------------------------
// Phase 0: W [N=1024][K=1024] f32 (row = output feature) -> bf16 swizzled
// B-operand blocks: out[((kt*NT + nt)*32 + lane)*16 + e] = W[n][k]
//   n = nt*16 + (lane & 15)
//   k = kt*32 + (lane&16 ? 8 : 0) + (e&7) + (e&8 ? 16 : 0)   (matches A frag K order)
// ---------------------------------------------------------------------------
__global__ __launch_bounds__(256) void swizzle_w(const float* __restrict__ W,
                                                 __bf16* __restrict__ out) {
  int idx = blockIdx.x * 256 + threadIdx.x;       // 0 .. DH*DH-1 (exact)
  int e   = idx & 15;
  int l   = (idx >> 4) & 31;
  int rem = idx >> 9;                              // kt*NT + nt
  int nt  = rem & (NT - 1);
  int kt  = rem >> 6;                              // log2(NT) = 6
  int n   = nt * 16 + (l & 15);
  int k   = kt * 32 + ((l & 16) ? 8 : 0) + (e & 7) + ((e & 8) ? 16 : 0);
  out[idx] = (__bf16)W[(size_t)n * DH + k];
}

// ---------------------------------------------------------------------------
// Phase 1: xproj[m][n] = sum_k x[m][k] * Wxh[n][k] + bh[n], stored bf16.
// Block = 128 threads (4 waves), block tile 64x64, each wave 32x32 (2x2 tiles).
// A (x, f32 row-major) converted to bf16 in-register.
// ---------------------------------------------------------------------------
__global__ __launch_bounds__(128) void gemm_xproj(
    const float*  __restrict__ x,      // [MTOT][DIN] f32
    const __bf16* __restrict__ Bsw,    // swizzled Wxh, [KT][NT][32][16]
    const float*  __restrict__ bh,     // [DH]
    __bf16*       __restrict__ xproj)  // [MTOT][DH] bf16
{
  const int wave = threadIdx.x >> 5;
  const int lane = threadIdx.x & 31;

  const int m0 = blockIdx.x * 64 + (wave >> 1) * 32;   // 32-row wave tile
  const int n0 = blockIdx.y * 64 + (wave & 1) * 32;    // 32-col wave tile
  const int nt0 = n0 >> 4;

  const int rowA   = lane & 15;                 // A frag row within tile
  const int kshift = (lane & 16) ? 8 : 0;       // A frag K sub-offset
  const int mlb    = (lane & 16) ? 8 : 0;       // D row base
  const int col    = lane & 15;                 // D column

  v8f acc00 = {}, acc01 = {}, acc10 = {}, acc11 = {};

  for (int kt = 0; kt < KT; ++kt) {
    const int kb = kt * 32 + kshift;

    // ---- A fragments (f32 -> bf16 on the fly) --------------------------
    v16bf a0, a1;
    {
      const float* p = x + (size_t)(m0 + rowA) * DIN + kb;
      v4f f0 = *(const v4f*)(p);        // K +0..3
      v4f f1 = *(const v4f*)(p + 4);    // K +4..7
      v4f f2 = *(const v4f*)(p + 16);   // K +16..19
      v4f f3 = *(const v4f*)(p + 20);   // K +20..23
#pragma unroll
      for (int i = 0; i < 4; ++i) {
        a0[i]      = (__bf16)f0[i];
        a0[4 + i]  = (__bf16)f1[i];
        a0[8 + i]  = (__bf16)f2[i];
        a0[12 + i] = (__bf16)f3[i];
      }
    }
    {
      const float* p = x + (size_t)(m0 + 16 + rowA) * DIN + kb;
      v4f f0 = *(const v4f*)(p);
      v4f f1 = *(const v4f*)(p + 4);
      v4f f2 = *(const v4f*)(p + 16);
      v4f f3 = *(const v4f*)(p + 20);
#pragma unroll
      for (int i = 0; i < 4; ++i) {
        a1[i]      = (__bf16)f0[i];
        a1[4 + i]  = (__bf16)f1[i];
        a1[8 + i]  = (__bf16)f2[i];
        a1[12 + i] = (__bf16)f3[i];
      }
    }

    // ---- B fragments (pre-swizzled: contiguous 32B per lane) -----------
    const v16bf b0 = *(const v16bf*)(Bsw + (((size_t)kt * NT + nt0)     * 32 + lane) * 16);
    const v16bf b1 = *(const v16bf*)(Bsw + (((size_t)kt * NT + nt0 + 1) * 32 + lane) * 16);

    acc00 = wmma_bf16(a0, b0, acc00);
    acc01 = wmma_bf16(a0, b1, acc01);
    acc10 = wmma_bf16(a1, b0, acc10);
    acc11 = wmma_bf16(a1, b1, acc11);
  }

  // ---- epilogue: + bias, store bf16 (D layout: lane=col, vgpr=row) -----
  const float bias0 = bh[n0 + col];
  const float bias1 = bh[n0 + 16 + col];
#pragma unroll
  for (int v = 0; v < 8; ++v) {
    const int m = mlb + v;
    xproj[(size_t)(m0 + m)      * DH + n0 + col]      = (__bf16)(acc00[v] + bias0);
    xproj[(size_t)(m0 + m)      * DH + n0 + 16 + col] = (__bf16)(acc01[v] + bias1);
    xproj[(size_t)(m0 + 16 + m) * DH + n0 + col]      = (__bf16)(acc10[v] + bias0);
    xproj[(size_t)(m0 + 16 + m) * DH + n0 + 16 + col] = (__bf16)(acc11[v] + bias1);
  }
}

// ---------------------------------------------------------------------------
// Phase 2: persistent recurrence. 4 blocks x 512 threads (16 waves).
// Each block owns 16 batch rows; h (16x1024 bf16) lives in LDS with +8
// element row padding (row stride 2064B = 4-bank step -> conflict-free
// ds_read_b128 A-fragment loads). Each wave owns 4 n-tiles (64 columns).
// ---------------------------------------------------------------------------
constexpr int ROWSTRIDE = DH + 8;   // bf16 elements per padded LDS row

__global__ __launch_bounds__(512) void rnn_scan(
    const __bf16* __restrict__ xproj,  // [T*B][DH] bf16
    const __bf16* __restrict__ Wsw,    // swizzled Whh, [KT][NT][32][16]
    float*        __restrict__ out,    // [T*B][DH] f32
    float*        __restrict__ hlast)  // [B][DH] f32
{
  __shared__ __align__(16) __bf16 hbuf[16 * ROWSTRIDE];   // 33 KB

  const int tid  = threadIdx.x;
  const int wave = tid >> 5;
  const int lane = tid & 31;
  const int b0   = blockIdx.x * 16;    // batch chunk base

  // h0 = 0
  for (int i = tid; i < 16 * ROWSTRIDE; i += 512) hbuf[i] = (__bf16)0.0f;
  __syncthreads();

  const int nt0    = wave * 4;               // 4 n-tiles per wave
  const int rowA   = lane & 15;
  const int kshift = (lane & 16) ? 8 : 0;
  const int mlb    = (lane & 16) ? 8 : 0;
  const int col    = lane & 15;

  for (int t = 0; t < T_STEPS; ++t) {
    // ---- init accumulators from xproj[t] (D layout gather) -------------
    const __bf16* xp = xproj + ((size_t)t * BATCH + b0) * DH;
    v8f acc[4];
#pragma unroll
    for (int j = 0; j < 4; ++j) {
      const int n0 = (nt0 + j) * 16;
#pragma unroll
      for (int v = 0; v < 8; ++v)
        acc[j][v] = (float)xp[(size_t)(mlb + v) * DH + n0 + col];
    }

    // ---- prefetch xproj[t+1]: 512 threads x 64B = full 32KB tile -------
    {
      const int tn = (t + 1 < T_STEPS) ? (t + 1) : t;
      const __bf16* xpn = xproj + ((size_t)tn * BATCH + b0) * DH + (size_t)tid * 32;
      __builtin_prefetch(xpn, 0, 3);   // -> global_prefetch_b8
    }

    // ---- h @ Whh^T over K = 1024 ---------------------------------------
#pragma unroll 2
    for (int kt = 0; kt < KT; ++kt) {
      const int kb = kt * 32 + kshift;
      const __bf16* p = hbuf + rowA * ROWSTRIDE + kb;
      v8bf lo = *(const v8bf*)(p);        // K +0..7  (16B aligned)
      v8bf hi = *(const v8bf*)(p + 16);   // K +16..23
      v16bf a = __builtin_shufflevector(lo, hi,
          0, 1, 2, 3, 4, 5, 6, 7, 8, 9, 10, 11, 12, 13, 14, 15);
#pragma unroll
      for (int j = 0; j < 4; ++j) {
        const v16bf bb = *(const v16bf*)(
            Wsw + (((size_t)kt * NT + nt0 + j) * 32 + lane) * 16);
        acc[j] = wmma_bf16(a, bb, acc[j]);
      }
    }

    __syncthreads();   // all waves done READING h for step t

    // ---- epilogue: tanh, store out (f32), refresh h in LDS (bf16) ------
    const bool last = (t == T_STEPS - 1);
    float* orow = out + ((size_t)t * BATCH + b0) * DH;
#pragma unroll
    for (int j = 0; j < 4; ++j) {
      const int n0 = (nt0 + j) * 16;
#pragma unroll
      for (int v = 0; v < 8; ++v) {
        const int m = mlb + v;
        const float hv = tanhf(acc[j][v]);
        orow[(size_t)m * DH + n0 + col] = hv;
        hbuf[m * ROWSTRIDE + n0 + col]  = (__bf16)hv;
        if (last) hlast[(size_t)(b0 + m) * DH + n0 + col] = hv;
      }
    }

    __syncthreads();   // new h visible before step t+1 reads
  }
}

// ---------------------------------------------------------------------------
extern "C" void kernel_launch(void* const* d_in, const int* in_sizes, int n_in,
                              void* d_out, int out_size, void* d_ws, size_t ws_size,
                              hipStream_t stream) {
  const float* x   = (const float*)d_in[0];   // [T, B, IN]
  const float* Wxh = (const float*)d_in[1];   // [H, IN]
  const float* Whh = (const float*)d_in[2];   // [H, H]
  const float* bh  = (const float*)d_in[3];   // [H]

  float* out   = (float*)d_out;                         // [T, B, H]
  float* hlast = out + (size_t)MTOT * DH;               // [B, H]

  // Workspace layout (total ~68 MB): Wxh_sw(2MB) | Whh_sw(2MB) | xproj(64MB)
  __bf16* wxh_sw = (__bf16*)d_ws;
  __bf16* whh_sw = wxh_sw + (size_t)DH * DIN;
  __bf16* xproj  = whh_sw + (size_t)DH * DH;

  swizzle_w<<<(DH * DIN) / 256, 256, 0, stream>>>(Wxh, wxh_sw);
  swizzle_w<<<(DH * DH)  / 256, 256, 0, stream>>>(Whh, whh_sw);

  dim3 g1(MTOT / 64, DH / 64);
  gemm_xproj<<<g1, 128, 0, stream>>>(x, wxh_sw, bh, xproj);

  rnn_scan<<<BATCH / 16, 512, 0, stream>>>(xproj, whh_sw, out, hlast);
}
